// RNNImputerModel_66391604462223
// MI455X (gfx1250) — matrix operand order repaired and checked by reading, not compiled, not run on hardware
//
#include <hip/hip_runtime.h>
#include <hip/hip_bf16.h>

// ---------------- problem constants ----------------
#define B_    64
#define S_    512
#define D_    207          // node features
#define H_    256          // hidden
#define DIN_  414          // 2*D  (concat mask)
#define KCAT  670          // D + D + H packed input for fused K
#define NSTEP (S_ - 1)     // 511 recurrent steps

#define NWG   8
#define TPB   256          // 8 wave32 per block -> 64 waves total

// fragment tiling
#define NT_G  48           // 768 gate columns / 16
#define KT_I  13           // ceil(414/32)
#define KT_H  8            // 256/32
#define NT_R  13           // ceil(207/16)
#define KT_R  8
#define FR_IH (NT_G * KT_I)          // 624
#define FR_HH (NT_G * KT_H)          // 384
#define FR_RO (NT_R * KT_R)          // 104
#define FR_TOT (FR_IH + FR_HH + FR_RO)

typedef __attribute__((ext_vector_type(16))) __bf16       v16bf;
typedef __attribute__((ext_vector_type(8)))  float        v8f;
typedef __attribute__((ext_vector_type(8)))  unsigned int v8u;

static __device__ inline v16bf as_bf(v8u u) {
    union { v8u u; v16bf b; } c; c.u = u; return c.b;
}
static __device__ inline unsigned f2bf(float f) {   // round-to-nearest-even
    unsigned u = __float_as_uint(f);
    return (u + 0x7FFFu + ((u >> 16) & 1u)) >> 16;
}
static __device__ inline float fsigmoid(float v) { return 1.f / (1.f + __expf(-v)); }
static __device__ inline float ftanh(float v)    { return 1.f - 2.f / (__expf(2.f * v) + 1.f); }

// A-fragment (16x32 bf16) gather: per lane, 8 dwords of K-pairs.
// lane<16: K = 0..7,16..23 (pairs v&3, group v>>2 adds 16); lane>=16: +8.
static __device__ inline v8u load_a(const unsigned* row, int pairbase, int kh) {
    const unsigned* p = row + pairbase + kh * 4;
    v8u r;
    r[0] = p[0]; r[1] = p[1]; r[2] = p[2];  r[3] = p[3];
    r[4] = p[8]; r[5] = p[9]; r[6] = p[10]; r[7] = p[11];
    return r;
}

// ---------------- pre-pass: pack weights into WMMA B-fragment layout ----------------
// B-frag (32x16 bf16): lane = 16*((K>>3)&1) + N;  dword v holds K pair (k, k+1), k = 8*(lane>>4) + 2*(v&3) + 16*(v>>2).
__global__ void pack_weights(const float* __restrict__ Wih,
                             const float* __restrict__ Whh,
                             const float* __restrict__ Wro,
                             unsigned*    __restrict__ frags) {
    int frag = blockIdx.x;
    int lane = threadIdx.x & 31, v = threadIdx.x >> 5;
    const float* src; int nt, kt, Nreal, Kreal, ldk;
    if (frag < FR_IH)             { src = Wih; nt = frag / KT_I; kt = frag % KT_I; Nreal = 3 * H_; Kreal = DIN_; ldk = DIN_; }
    else if (frag < FR_IH + FR_HH){ int r = frag - FR_IH; src = Whh; nt = r / KT_H; kt = r % KT_H; Nreal = 3 * H_; Kreal = H_; ldk = H_; }
    else                          { int r = frag - FR_IH - FR_HH; src = Wro; nt = r / KT_R; kt = r % KT_R; Nreal = D_; Kreal = H_; ldk = H_; }
    int n  = nt * 16 + (lane & 15);
    int k0 = kt * 32 + (lane >> 4) * 8 + (v & 3) * 2 + (v >> 2) * 16;
    float f0 = 0.f, f1 = 0.f;
    if (n < Nreal) {
        if (k0     < Kreal) f0 = src[n * ldk + k0];
        if (k0 + 1 < Kreal) f1 = src[n * ldk + k0 + 1];
    }
    frags[frag * 256 + lane * 8 + v] = f2bf(f0) | (f2bf(f1) << 16);
}

// ---------------- pre-pass: init state, barrier counter, out[:,0,:] ----------------
__global__ void init_state(const float* __restrict__ b_ro,
                           float* __restrict__ out,
                           float* __restrict__ xhat,
                           unsigned* __restrict__ hbf,   // 2 buffers of 64x128 dwords
                           unsigned* __restrict__ cnt) {
    int i = blockIdx.x * blockDim.x + threadIdx.x;
    int stride = gridDim.x * blockDim.x;
    if (i == 0) *cnt = 0u;
    for (int j = i; j < 2 * B_ * 128; j += stride) hbf[j] = 0u;
    for (int j = i; j < B_ * 208; j += stride) {
        int b = j / 208, d = j % 208;
        float v = (d < D_) ? b_ro[d] : 0.f;
        xhat[j] = v;
        if (d < D_) out[(size_t)b * S_ * D_ + d] = v;   // preds[0] = readout(h0=0)
    }
}

// ---------------- device-scope spin barrier ----------------
static __device__ inline void gridbar(unsigned* cnt, unsigned target) {
    __threadfence();
    __syncthreads();
    if (threadIdx.x == 0) {
        __hip_atomic_fetch_add(cnt, 1u, __ATOMIC_ACQ_REL, __HIP_MEMORY_SCOPE_AGENT);
        while (__hip_atomic_load(cnt, __ATOMIC_ACQUIRE, __HIP_MEMORY_SCOPE_AGENT) < target)
            __builtin_amdgcn_s_sleep(2);
    }
    __syncthreads();
}

// ---------------- persistent recurrent kernel ----------------
__launch_bounds__(TPB, 1)
__global__ void rnn_persistent(const float* __restrict__ x,
                               const float* __restrict__ mask,
                               const float* __restrict__ b_ih,
                               const float* __restrict__ b_hh,
                               const float* __restrict__ b_ro,
                               const unsigned* __restrict__ frags,
                               unsigned* __restrict__ hbf,    // double-buffered bf16 h
                               float* __restrict__ xhat,
                               float* __restrict__ out,
                               unsigned* __restrict__ cnt) {
    __shared__ unsigned lds_x[16 * 336];   // this block's 16 batch rows of x_cat (bf16 pairs)

    const int tid  = threadIdx.x;
    const int lane = tid & 31;
    const int w    = blockIdx.x * 8 + (tid >> 5);   // global wave id 0..63
    const int mt   = w >> 4;                        // batch tile 0..3
    const int ht   = w & 15;                        // hidden tile 0..15
    const int blk_mt = blockIdx.x >> 1;             // == mt for all waves of this block
    const int kh = lane >> 4, ln = lane & 15;

    const unsigned* WihF = frags;
    const unsigned* WhhF = frags + (size_t)FR_IH * 256;
    const unsigned* WroF = frags + (size_t)(FR_IH + FR_HH) * 256;

    // per-lane gate biases (hidden unit j)
    const int j = ht * 16 + ln;
    const float bir = b_ih[j],          bhr = b_hh[j];
    const float biz = b_ih[H_ + j],     bhz = b_hh[H_ + j];
    const float bin_= b_ih[2 * H_ + j], bhn = b_hh[2 * H_ + j];

    // readout tile assignment (52 of 64 waves)
    const bool do_ro = (w < 4 * NT_R);
    const int  mt_r  = w / NT_R, nt_r = w % NT_R;
    const int  n_d   = nt_r * 16 + ln;
    const float bro_n = (do_ro && n_d < D_) ? b_ro[n_d] : 0.f;

    v8f hreg = {0, 0, 0, 0, 0, 0, 0, 0};   // resident fp32 h tile, h0 = 0
    const v8f vzero = {0, 0, 0, 0, 0, 0, 0, 0};
    unsigned epoch = 0;

    for (int t = 0; t < NSTEP; ++t) {
        const unsigned* hbf_r = hbf + (size_t)(t & 1) * (B_ * 128);
        unsigned*       hbf_w = hbf + (size_t)((t + 1) & 1) * (B_ * 128);

        // ---- phase 1: build x_cat = [x_p | mask | h] for this block's 16 rows ----
        for (int idx = tid; idx < 16 * 336; idx += TPB) {
            int mrow = blk_mt * 16 + (idx / 336);
            int kp   = idx % 336;
            unsigned packed = 0;
            #pragma unroll
            for (int half = 0; half < 2; ++half) {
                int k = kp * 2 + half;
                unsigned val = 0;
                if (k < D_) {
                    size_t gi = ((size_t)mrow * S_ + t) * D_ + k;
                    float mv = mask[gi];
                    float xv = (mv > 0.5f) ? x[gi] : xhat[mrow * 208 + k];
                    val = f2bf(xv);
                } else if (k < 2 * D_) {
                    val = f2bf(mask[((size_t)mrow * S_ + t) * D_ + (k - D_)]);
                } else if (k < KCAT) {
                    val = ((const unsigned short*)hbf_r)[mrow * H_ + (k - 2 * D_)];
                }
                packed |= val << (16 * half);
            }
            lds_x[idx] = packed;
        }
        __syncthreads();

        // ---- phase 2: fused gate GEMMs (gi over K=414, gh over K=256) ----
        v8f gi0 = vzero, gi1 = vzero, gi2 = vzero;
        v8f gh0 = vzero, gh1 = vzero, gh2 = vzero;
        const unsigned* arow = lds_x + ln * 336;
        #pragma unroll
        for (int kt = 0; kt < KT_I; ++kt) {
            v16bf a = as_bf(load_a(arow, kt * 16, kh));
            const v8u* b0 = ((const v8u*)WihF) + ((0 * 16 + ht) * KT_I + kt) * 32 + lane;
            const v8u* b1 = ((const v8u*)WihF) + ((1 * 16 + ht) * KT_I + kt) * 32 + lane;
            const v8u* b2 = ((const v8u*)WihF) + ((2 * 16 + ht) * KT_I + kt) * 32 + lane;
            gi0 = __builtin_amdgcn_wmma_f32_16x16x32_bf16(false, a, false, as_bf(*b0), (short)0, gi0, false, false);
            gi1 = __builtin_amdgcn_wmma_f32_16x16x32_bf16(false, a, false, as_bf(*b1), (short)0, gi1, false, false);
            gi2 = __builtin_amdgcn_wmma_f32_16x16x32_bf16(false, a, false, as_bf(*b2), (short)0, gi2, false, false);
        }
        #pragma unroll
        for (int kt = 0; kt < KT_H; ++kt) {
            v16bf a = as_bf(load_a(arow, D_ + kt * 16, kh));   // pair offset 207 == byte offset 414*2
            const v8u* b0 = ((const v8u*)WhhF) + ((0 * 16 + ht) * KT_H + kt) * 32 + lane;
            const v8u* b1 = ((const v8u*)WhhF) + ((1 * 16 + ht) * KT_H + kt) * 32 + lane;
            const v8u* b2 = ((const v8u*)WhhF) + ((2 * 16 + ht) * KT_H + kt) * 32 + lane;
            gh0 = __builtin_amdgcn_wmma_f32_16x16x32_bf16(false, a, false, as_bf(*b0), (short)0, gh0, false, false);
            gh1 = __builtin_amdgcn_wmma_f32_16x16x32_bf16(false, a, false, as_bf(*b1), (short)0, gh1, false, false);
            gh2 = __builtin_amdgcn_wmma_f32_16x16x32_bf16(false, a, false, as_bf(*b2), (short)0, gh2, false, false);
        }

        // ---- phase 3: gate nonlinearities + resident h update ----
        #pragma unroll
        for (int p = 0; p < 8; ++p) {
            float r  = fsigmoid(gi0[p] + bir + gh0[p] + bhr);
            float z  = fsigmoid(gi1[p] + biz + gh1[p] + bhz);
            float n  = ftanh((gi2[p] + bin_) + r * (gh2[p] + bhn));
            float hn = (1.f - z) * n + z * hreg[p];
            hreg[p] = hn;
            int mrow = mt * 16 + p + kh * 8;                  // C/D layout: VGPR p -> M=p (+8 for hi half)
            ((unsigned short*)hbf_w)[mrow * H_ + j] = (unsigned short)f2bf(hn);
        }
        epoch += NWG;
        gridbar(cnt, epoch);          // h_{t+1} globally visible

        // ---- phase 4: readout x_hat = h_new @ W_ro^T + b_ro ----
        if (do_ro) {
            v8f acc = vzero;
            const unsigned* hrow = hbf_w + (mt_r * 16 + ln) * 128;
            #pragma unroll
            for (int kt = 0; kt < KT_R; ++kt) {
                v16bf a = as_bf(load_a(hrow, kt * 16, kh));
                const v8u* bp = ((const v8u*)WroF) + (nt_r * KT_R + kt) * 32 + lane;
                acc = __builtin_amdgcn_wmma_f32_16x16x32_bf16(false, a, false, as_bf(*bp), (short)0, acc, false, false);
            }
            if (n_d < D_) {
                #pragma unroll
                for (int p = 0; p < 8; ++p) {
                    int mrow = mt_r * 16 + p + kh * 8;
                    float v = acc[p] + bro_n;
                    xhat[mrow * 208 + n_d] = v;
                    out[((size_t)mrow * S_ + (t + 1)) * D_ + n_d] = v;
                }
            }
        }
        epoch += NWG;
        gridbar(cnt, epoch);          // x_hat_{t+1} globally visible
    }
}

// ---------------- host launcher ----------------
extern "C" void kernel_launch(void* const* d_in, const int* in_sizes, int n_in,
                              void* d_out, int out_size, void* d_ws, size_t ws_size,
                              hipStream_t stream) {
    const float* x    = (const float*)d_in[0];
    const float* mask = (const float*)d_in[1];
    const float* Wih  = (const float*)d_in[2];
    const float* Whh  = (const float*)d_in[3];
    const float* bih  = (const float*)d_in[4];
    const float* bhh  = (const float*)d_in[5];
    const float* Wro  = (const float*)d_in[6];
    const float* bro  = (const float*)d_in[7];
    float* out = (float*)d_out;

    char* ws = (char*)d_ws;
    unsigned* cnt   = (unsigned*)ws;                                       // 256 B
    unsigned* frags = (unsigned*)(ws + 256);                               // FR_TOT KiB
    unsigned* hbf   = (unsigned*)(ws + 256 + (size_t)FR_TOT * 1024);       // 2 x 32 KiB
    float*    xhat  = (float*)(ws + 256 + (size_t)FR_TOT * 1024 + 2 * B_ * H_ * 2);  // 64x208 f32

    pack_weights<<<FR_TOT, 256, 0, stream>>>(Wih, Whh, Wro, frags);
    init_state<<<64, 256, 0, stream>>>(bro, out, xhat, hbf, cnt);
    rnn_persistent<<<NWG, TPB, 0, stream>>>(x, mask, bih, bhh, bro, frags, hbf, xhat, out, cnt);
}
